// AttnDecoderRNN_88923002896720
// MI455X (gfx1250) — matrix operand order repaired and checked by reading, not compiled, not run on hardware
//
#include <hip/hip_runtime.h>
#include <hip/hip_bf16.h>
#include <stdint.h>

// Problem constants (from reference)
#define Bc    64
#define NPIXc 49
#define LARTc 180
#define LCAPc 31
#define Tc    30
#define Vc    20000
#define Dc    512

typedef unsigned short bf16_t;
typedef __attribute__((ext_vector_type(16))) __bf16 v16bf;
typedef __attribute__((ext_vector_type(8)))  float  v8f;

// ---------------- device helpers ----------------
static __device__ __forceinline__ bf16_t f2bf(float x) {
  unsigned int u = __float_as_uint(x);
  u += 0x7FFFu + ((u >> 16) & 1u);      // round-to-nearest-even
  return (bf16_t)(u >> 16);
}
static __device__ __forceinline__ float bf2f(bf16_t h) {
  return __uint_as_float(((unsigned int)h) << 16);
}
static __device__ __forceinline__ float sigmf(float x) { return 1.0f / (1.0f + expf(-x)); }

static __device__ __forceinline__ float wave_sum(float v) {
#pragma unroll
  for (int o = 16; o > 0; o >>= 1) v += __shfl_xor(v, o, 32);
  return v;
}
static __device__ __forceinline__ float wave_max(float v) {
#pragma unroll
  for (int o = 16; o > 0; o >>= 1) v = fmaxf(v, __shfl_xor(v, o, 32));
  return v;
}

union FragU { v16bf v; uint4 q[2]; };

// A fragment: 16x32 bf16, per ISA layout (row = lane&15, K split +0/+8, halves at +0/+16)
static __device__ __forceinline__ v16bf frag_a(const bf16_t* p) {
  FragU u;
  u.q[0] = *(const uint4*)(p);
  u.q[1] = *(const uint4*)(p + 16);
  return u.v;
}
// B fragment: 32x16 bf16 (col = lane&15, K split +0/+8, halves at +0/+16)
static __device__ __forceinline__ v16bf frag_b(const bf16_t* p) {
  FragU u;
  u.q[0] = *(const uint4*)(p);
  u.q[1] = *(const uint4*)(p + 16);
  return u.v;
}
static __device__ __forceinline__ v8f wmma_bf16(v16bf a, v16bf b, v8f c) {
  return __builtin_amdgcn_wmma_f32_16x16x32_bf16(false, a, false, b, (short)0, c, false, false);
}

// ---------------- generic GEMM: C[M,N] = act(A[M,K] @ W[N,K]^T + bias) ----------------
// Two-stage software-pipelined K loop (K must be a multiple of 64, true for all call sites).
// NSUB = number of 16-col sub-tiles per wave (compile-time, 1..4).
// act: 0=none, 1=relu, 2=tanh.  Cf (f32) and/or Cb (bf16) outputs, either may be null.
template <int NSUB>
__global__ __launch_bounds__(256) void gemm_bf16_kernel(
    const bf16_t* __restrict__ A, long lda,
    const bf16_t* __restrict__ W, long ldw,
    const float* __restrict__ bias,
    float* __restrict__ Cf, long ldc,
    bf16_t* __restrict__ Cb, long ldcb,
    int M, int colBase, int ngrp, int K, int act)
{
  int wave = blockIdx.x * (blockDim.x >> 5) + (threadIdx.x >> 5);
  int lane = threadIdx.x & 31;
  int mt = wave / ngrp, ng = wave % ngrp;
  int row0 = mt * 16;
  if (row0 >= M) return;
  int col0 = colBase + ng * (NSUB * 16);

  const bf16_t* pa = A + (long)(row0 + (lane & 15)) * lda + ((lane >> 4) << 3);
  const bf16_t* pw[NSUB];
#pragma unroll
  for (int j = 0; j < NSUB; ++j)
    pw[j] = W + (long)(col0 + j * 16 + (lane & 15)) * ldw + ((lane >> 4) << 3);

  v8f acc[NSUB] = {};
  v16bf a0, a1, b0[NSUB], b1[NSUB];
  a0 = frag_a(pa);
#pragma unroll
  for (int j = 0; j < NSUB; ++j) b0[j] = frag_b(pw[j]);

  for (int k0 = 0; k0 < K; k0 += 64) {
    // stage-1 loads (k0+32) in flight while stage-0 computes
    a1 = frag_a(pa + 32);
#pragma unroll
    for (int j = 0; j < NSUB; ++j) b1[j] = frag_b(pw[j] + 32);
#pragma unroll
    for (int j = 0; j < NSUB; ++j) acc[j] = wmma_bf16(a0, b0[j], acc[j]);
    // stage-0 loads for next iteration (uniform guard; avoids OOB on last iter)
    if (k0 + 64 < K) {
      a0 = frag_a(pa + 64);
#pragma unroll
      for (int j = 0; j < NSUB; ++j) b0[j] = frag_b(pw[j] + 64);
    }
#pragma unroll
    for (int j = 0; j < NSUB; ++j) acc[j] = wmma_bf16(a1, b1[j], acc[j]);
    pa += 64;
#pragma unroll
    for (int j = 0; j < NSUB; ++j) pw[j] += 64;
  }

  int n = lane & 15, rsel = lane >> 4;
#pragma unroll
  for (int j = 0; j < NSUB; ++j) {
    int col = col0 + j * 16 + n;
    float bv = bias ? bias[col] : 0.0f;
#pragma unroll
    for (int r = 0; r < 8; ++r) {
      int row = row0 + rsel * 8 + r;
      float v = acc[j][r] + bv;
      if (act == 1) v = fmaxf(v, 0.0f);
      else if (act == 2) v = tanhf(v);
      if (Cf) Cf[(long)row * ldc + col] = v;
      if (Cb) Cb[(long)row * ldcb + col] = f2bf(v);
    }
  }
}

// ---------------- shared WMMA body for 4-gate recurrent kernels ----------------
// Accumulates acc[g] += A[16xK] @ Wgate_g[K x 16]^T with 2-stage pipelining.
static __device__ __forceinline__ void gate_gemm_acc(
    const bf16_t* __restrict__ Abase, long lda,
    const bf16_t* __restrict__ Wbase, long ldw,
    int row0, int h0, int lane, int K, v8f acc[4])
{
  const bf16_t* pa = Abase + (long)(row0 + (lane & 15)) * lda + ((lane >> 4) << 3);
  const bf16_t* pw[4];
#pragma unroll
  for (int g = 0; g < 4; ++g)
    pw[g] = Wbase + (long)(g * 512 + h0 + (lane & 15)) * ldw + ((lane >> 4) << 3);

  v16bf a0, a1, b0[4], b1[4];
  a0 = frag_a(pa);
#pragma unroll
  for (int g = 0; g < 4; ++g) b0[g] = frag_b(pw[g]);

  for (int k0 = 0; k0 < K; k0 += 64) {
    a1 = frag_a(pa + 32);
#pragma unroll
    for (int g = 0; g < 4; ++g) b1[g] = frag_b(pw[g] + 32);
#pragma unroll
    for (int g = 0; g < 4; ++g) acc[g] = wmma_bf16(a0, b0[g], acc[g]);
    if (k0 + 64 < K) {
      a0 = frag_a(pa + 64);
#pragma unroll
      for (int g = 0; g < 4; ++g) b0[g] = frag_b(pw[g] + 64);
    }
#pragma unroll
    for (int g = 0; g < 4; ++g) acc[g] = wmma_bf16(a1, b1[g], acc[g]);
    pa += 64;
#pragma unroll
    for (int g = 0; g < 4; ++g) pw[g] += 64;
  }
}

// ---------------- fused bidirectional-LSTM step ----------------
// g = XW[:,t,:] + h_prev @ Whh^T ; gate update; writes c_next(f32), h_next(bf16), hiddens slice(bf16)
__global__ __launch_bounds__(256) void lstm_step_kernel(
    const bf16_t* __restrict__ hprev, const float* __restrict__ cprev,
    const bf16_t* __restrict__ Whh,   const float* __restrict__ XW,
    int t, int L,
    float* __restrict__ cnext, bf16_t* __restrict__ hnext,
    bf16_t* __restrict__ hiddens, int hoff)
{
  int wave = blockIdx.x * (blockDim.x >> 5) + (threadIdx.x >> 5);
  int lane = threadIdx.x & 31;
  int mt = wave >> 5;          // 4 row tiles (B=64)
  int ht = wave & 31;          // 32 col tiles (H=512)
  int row0 = mt * 16, h0 = ht * 16;

  v8f acc[4] = {};
  gate_gemm_acc(hprev, 512, Whh, 512, row0, h0, lane, 512, acc);

  int n = lane & 15, rsel = lane >> 4;
  int col = h0 + n;
#pragma unroll
  for (int r = 0; r < 8; ++r) {
    int b = row0 + rsel * 8 + r;
    long xrow = ((long)b * L + t) * 2048;
    float gi = acc[0][r] + XW[xrow + col];
    float gf = acc[1][r] + XW[xrow + 512 + col];
    float gg = acc[2][r] + XW[xrow + 1024 + col];
    float go = acc[3][r] + XW[xrow + 1536 + col];
    float c = sigmf(gf) * cprev[b * 512 + col] + sigmf(gi) * tanhf(gg);
    float h = sigmf(go) * tanhf(c);
    cnext[b * 512 + col] = c;
    bf16_t hb = f2bf(h);
    hnext[b * 512 + col] = hb;
    hiddens[((long)b * L + t) * 1024 + hoff + col] = hb;
  }
}

// ---------------- fused decoder LSTMCell ----------------
// g = cellin @ Wih^T + bias + h_att @ Whh^T; writes h1->f1in[:, :512], c1->f2in[:, :512] (bf16)
__global__ __launch_bounds__(256) void cell_step_kernel(
    const bf16_t* __restrict__ cellin, const bf16_t* __restrict__ hatt,
    const bf16_t* __restrict__ Wih,    const bf16_t* __restrict__ Whh,
    const float* __restrict__ bias,    const float* __restrict__ catt,
    bf16_t* __restrict__ f1in, bf16_t* __restrict__ f2in)
{
  int wave = blockIdx.x * (blockDim.x >> 5) + (threadIdx.x >> 5);
  int lane = threadIdx.x & 31;
  int mt = wave >> 5, ht = wave & 31;
  int row0 = mt * 16, h0 = ht * 16;

  v8f acc[4] = {};
  gate_gemm_acc(cellin, 1024, Wih, 1024, row0, h0, lane, 1024, acc);
  gate_gemm_acc(hatt,   512,  Whh, 512,  row0, h0, lane, 512,  acc);

  int n = lane & 15, rsel = lane >> 4;
  int col = h0 + n;
#pragma unroll
  for (int r = 0; r < 8; ++r) {
    int b = row0 + rsel * 8 + r;
    float gi = acc[0][r] + bias[col];
    float gf = acc[1][r] + bias[512 + col];
    float gg = acc[2][r] + bias[1024 + col];
    float go = acc[3][r] + bias[1536 + col];
    float c1 = sigmf(gf) * catt[b * 512 + col] + sigmf(gi) * tanhf(gg);
    float h1 = sigmf(go) * tanhf(c1);
    f1in[b * 1024 + col] = f2bf(h1);
    f2in[b * 1024 + col] = f2bf(c1);
  }
}

// ---------------- attention score kernels (one wave per (b,l)) ----------------
__global__ void score1_kernel(const float* __restrict__ encH, const float* __restrict__ encI,
                              const float* __restrict__ dec, const float* __restrict__ fw,
                              const float* __restrict__ fb, float* __restrict__ s1, int rows)
{
  int wave = blockIdx.x * (blockDim.x >> 5) + (threadIdx.x >> 5);
  int lane = threadIdx.x & 31;
  if (wave >= rows) return;
  int b = wave / 229, l = wave % 229;
  const float* src = (l < 180) ? (encH + ((long)b * 180 + l) * 512)
                               : (encI + ((long)b * 49 + (l - 180)) * 512);
  const float* dv = dec + b * 512;
  float sum = 0.0f;
  for (int k = lane; k < 512; k += 32) sum += tanhf(src[k] + dv[k]) * fw[k];
  sum = wave_sum(sum);
  if (lane == 0) s1[wave] = sum + fb[0];
}

__global__ void score2_kernel(const float* __restrict__ enc2, const float* __restrict__ dec,
                              const float* __restrict__ fw, const float* __restrict__ fb,
                              const int* __restrict__ mask, float* __restrict__ s2, int rows)
{
  int wave = blockIdx.x * (blockDim.x >> 5) + (threadIdx.x >> 5);
  int lane = threadIdx.x & 31;
  if (wave >= rows) return;
  int b = wave / 180;
  const float* src = enc2 + (long)wave * 512;
  const float* dv = dec + b * 512;
  float sum = 0.0f;
  for (int k = lane; k < 512; k += 32) sum += tanhf(src[k] + dv[k]) * fw[k];
  sum = wave_sum(sum);
  if (lane == 0) s2[wave] = (mask[wave] == 0) ? -1e9f : (sum + fb[0]);
}

// ---------------- attention softmax + context (one block per batch row) ----------------
__global__ __launch_bounds__(256) void att1_kernel(
    const float* __restrict__ s1, const bf16_t* __restrict__ hid1,
    const bf16_t* __restrict__ enc1, const float* __restrict__ embs,
    int t, bf16_t* __restrict__ cellin)
{
  __shared__ float al[229];
  __shared__ float wred[8];
  int b = blockIdx.x, tid = threadIdx.x, lane = tid & 31, wid = tid >> 5;
  const float* srow = s1 + b * 229;
  float m = -1e30f;
  for (int l = tid; l < 229; l += 256) m = fmaxf(m, srow[l]);
  m = wave_max(m);
  if (lane == 0) wred[wid] = m;
  __syncthreads();
  float mm = fmaxf(fmaxf(fmaxf(wred[0], wred[1]), fmaxf(wred[2], wred[3])),
                   fmaxf(fmaxf(wred[4], wred[5]), fmaxf(wred[6], wred[7])));
  __syncthreads();
  float s = 0.0f;
  for (int l = tid; l < 229; l += 256) { float e = expf(srow[l] - mm); al[l] = e; s += e; }
  s = wave_sum(s);
  if (lane == 0) wred[wid] = s;
  __syncthreads();
  float inv = 1.0f / (wred[0] + wred[1] + wred[2] + wred[3] + wred[4] + wred[5] + wred[6] + wred[7]);
  for (int d = tid; d < 512; d += 256) {
    float acc = 0.0f;
    const bf16_t* ph = hid1 + (long)b * 180 * 512 + d;
    for (int l = 0; l < 180; ++l) acc += al[l] * bf2f(ph[(long)l * 512]);
    const bf16_t* pe = enc1 + (long)b * 49 * 512 + d;
    for (int l = 0; l < 49; ++l) acc += al[180 + l] * bf2f(pe[(long)l * 512]);
    acc *= inv;
    cellin[b * 1024 + d] = f2bf(acc);
    cellin[b * 1024 + 512 + d] = f2bf(embs[((long)b * Tc + t) * 512 + d]);
  }
}

__global__ __launch_bounds__(256) void att2_kernel(
    const float* __restrict__ s2, const bf16_t* __restrict__ hid2,
    const bf16_t* __restrict__ f1in, float* __restrict__ beta,
    float* __restrict__ guidef, bf16_t* __restrict__ fc1in)
{
  __shared__ float al[180];
  __shared__ float wred[8];
  int b = blockIdx.x, tid = threadIdx.x, lane = tid & 31, wid = tid >> 5;
  const float* srow = s2 + b * 180;
  float m = -1e30f;
  for (int l = tid; l < 180; l += 256) m = fmaxf(m, srow[l]);
  m = wave_max(m);
  if (lane == 0) wred[wid] = m;
  __syncthreads();
  float mm = fmaxf(fmaxf(fmaxf(wred[0], wred[1]), fmaxf(wred[2], wred[3])),
                   fmaxf(fmaxf(wred[4], wred[5]), fmaxf(wred[6], wred[7])));
  __syncthreads();
  float s = 0.0f;
  for (int l = tid; l < 180; l += 256) { float e = expf(srow[l] - mm); al[l] = e; s += e; }
  s = wave_sum(s);
  if (lane == 0) wred[wid] = s;
  __syncthreads();
  float inv = 1.0f / (wred[0] + wred[1] + wred[2] + wred[3] + wred[4] + wred[5] + wred[6] + wred[7]);
  for (int l = tid; l < 180; l += 256) beta[b * 180 + l] = al[l] * inv;
  for (int d = tid; d < 512; d += 256) {
    float acc = 0.0f;
    const bf16_t* ph = hid2 + (long)b * 180 * 512 + d;
    for (int l = 0; l < 180; ++l) acc += al[l] * bf2f(ph[(long)l * 512]);
    acc *= inv;
    guidef[b * 512 + d] = acc;
    fc1in[b * 1024 + d] = f2bf(acc);
    fc1in[b * 1024 + 512 + d] = f1in[b * 1024 + d];   // h1 (bf16 copy)
  }
}

// ---------------- pointer-generation gate: pg = sigmoid(pW . [emb_t, guide, h1] + pb) ----------------
__global__ void pgate_kernel(const float* __restrict__ embs, const float* __restrict__ guidef,
                             const bf16_t* __restrict__ f1in, const float* __restrict__ pW,
                             const float* __restrict__ pb, int t, float* __restrict__ pg)
{
  int wave = blockIdx.x * (blockDim.x >> 5) + (threadIdx.x >> 5);
  int lane = threadIdx.x & 31;
  if (wave >= Bc) return;
  int b = wave;
  float s = 0.0f;
  const float* e = embs + ((long)b * Tc + t) * 512;
  for (int k = lane; k < 512; k += 32) s += pW[k] * e[k];
  const float* g = guidef + b * 512;
  for (int k = lane; k < 512; k += 32) s += pW[512 + k] * g[k];
  const bf16_t* h = f1in + b * 1024;
  for (int k = lane; k < 512; k += 32) s += pW[1024 + k] * bf2f(h[k]);
  s = wave_sum(s);
  if (lane == 0) pg[b] = sigmf(s + pb[0]);
}

// ---------------- log-sum-exp over vocab (one block per batch row) ----------------
__global__ __launch_bounds__(256) void lse_kernel(const float* __restrict__ logits, float* __restrict__ lsm)
{
  __shared__ float wred[8];
  int b = blockIdx.x, tid = threadIdx.x, lane = tid & 31, wid = tid >> 5;
  const float* row = logits + (long)b * Vc;
  float m = -1e30f;
  for (int v = tid; v < Vc; v += 256) m = fmaxf(m, row[v]);
  m = wave_max(m);
  if (lane == 0) wred[wid] = m;
  __syncthreads();
  float mm = fmaxf(fmaxf(fmaxf(wred[0], wred[1]), fmaxf(wred[2], wred[3])),
                   fmaxf(fmaxf(wred[4], wred[5]), fmaxf(wred[6], wred[7])));
  __syncthreads();
  float s = 0.0f;
  for (int v = tid; v < Vc; v += 256) s += expf(row[v] - mm);
  s = wave_sum(s);
  if (lane == 0) wred[wid] = s;
  __syncthreads();
  if (tid == 0) {
    float ss = wred[0] + wred[1] + wred[2] + wred[3] + wred[4] + wred[5] + wred[6] + wred[7];
    lsm[b] = mm + logf(ss);
  }
}

// ---------------- final blend + copy-mechanism scatter ----------------
__global__ void final_kernel(const float* __restrict__ logits, const float* __restrict__ pg,
                             const float* __restrict__ lsm, const int* __restrict__ caplen,
                             int t, float* __restrict__ out)
{
  long i = (long)blockIdx.x * blockDim.x + threadIdx.x;
  if (i >= (long)Bc * Vc) return;
  int b = (int)(i / Vc), v = (int)(i % Vc);
  int act = t < (caplen[b] - 1);
  float val = act ? pg[b] * (logits[(long)b * Vc + v] - lsm[b]) : 0.0f;
  out[((long)b * Tc + t) * Vc + v] = val;
}

__global__ void scatter_kernel(const float* __restrict__ beta, const float* __restrict__ pg,
                               const int* __restrict__ art, const int* __restrict__ caplen,
                               int t, float* __restrict__ out)
{
  int i = blockIdx.x * blockDim.x + threadIdx.x;
  if (i >= Bc * LARTc) return;
  int b = i / LARTc;
  if (t >= caplen[b] - 1) return;
  int tok = art[i];
  atomicAdd(out + ((long)b * Tc + t) * Vc + tok, (1.0f - pg[b]) * beta[i]);
}

// ---------------- utility kernels ----------------
__global__ void cvt_kernel(const float* __restrict__ x, bf16_t* __restrict__ y, long n)
{
  long i = (long)blockIdx.x * blockDim.x + threadIdx.x;
  if (i < n) y[i] = f2bf(x[i]);
}
__global__ void zero_kernel(unsigned int* __restrict__ p, long n)
{
  long i = (long)blockIdx.x * blockDim.x + threadIdx.x;
  if (i < n) p[i] = 0u;
}
__global__ void gather_emb1_kernel(const float* __restrict__ emb1, const int* __restrict__ art,
                                   bf16_t* __restrict__ out, long n)
{
  long i = (long)blockIdx.x * blockDim.x + threadIdx.x;
  if (i >= n) return;
  long row = i >> 9; int d = (int)(i & 511);
  int tok = art[row];
  out[i] = f2bf(emb1[(long)tok * 512 + d]);
}
__global__ void gather_embs_kernel(const float* __restrict__ emb, const int* __restrict__ caps,
                                   float* __restrict__ out, long n)
{
  long i = (long)blockIdx.x * blockDim.x + threadIdx.x;
  if (i >= n) return;
  long row = i >> 9; int d = (int)(i & 511);
  int b = (int)(row / Tc), tt = (int)(row % Tc);
  int tok = caps[b * LCAPc + tt];
  out[i] = emb[(long)tok * 512 + d];
}
__global__ void shift_kernel(bf16_t* __restrict__ f1in, bf16_t* __restrict__ f2in)
{
  int i = blockIdx.x * blockDim.x + threadIdx.x;   // 64*512
  if (i >= Bc * 512) return;
  int b = i >> 9, d = i & 511;
  f1in[b * 1024 + 512 + d] = f1in[b * 1024 + d];
  f2in[b * 1024 + 512 + d] = f2in[b * 1024 + d];
}

// ---------------- host side ----------------
static void gemm(hipStream_t s, const bf16_t* A, long lda, const bf16_t* W, long ldw,
                 const float* bias, float* Cf, long ldc, bf16_t* Cb, long ldcb,
                 int M, int N, int K, int act)
{
  int ngrp = N >> 6;                       // full 64-wide column groups (NSUB=4)
  if (ngrp > 0) {
    int waves = (M / 16) * ngrp;
    gemm_bf16_kernel<4><<<dim3((waves + 7) / 8), dim3(256), 0, s>>>(
        A, lda, W, ldw, bias, Cf, ldc, Cb, ldcb, M, 0, ngrp, K, act);
  }
  int rem = (N - (ngrp << 6)) >> 4;        // leftover 16-col tiles (0..3)
  if (rem > 0) {
    int waves = M / 16;
    int colBase = ngrp << 6;
    int blocks = (waves + 7) / 8;
    switch (rem) {
      case 1: gemm_bf16_kernel<1><<<dim3(blocks), dim3(256), 0, s>>>(
                  A, lda, W, ldw, bias, Cf, ldc, Cb, ldcb, M, colBase, 1, K, act); break;
      case 2: gemm_bf16_kernel<2><<<dim3(blocks), dim3(256), 0, s>>>(
                  A, lda, W, ldw, bias, Cf, ldc, Cb, ldcb, M, colBase, 1, K, act); break;
      default: gemm_bf16_kernel<3><<<dim3(blocks), dim3(256), 0, s>>>(
                  A, lda, W, ldw, bias, Cf, ldc, Cb, ldcb, M, colBase, 1, K, act); break;
    }
  }
}

extern "C" void kernel_launch(void* const* d_in, const int* in_sizes, int n_in,
                              void* d_out, int out_size, void* d_ws, size_t ws_size,
                              hipStream_t stream)
{
  (void)in_sizes; (void)n_in; (void)out_size; (void)ws_size;

  const float* encoder_out = (const float*)d_in[0];
  const int*   captions    = (const int*)d_in[1];
  const int*   caplen      = (const int*)d_in[2];
  const int*   articles    = (const int*)d_in[3];
  const int*   mask        = (const int*)d_in[5];
  const float* emb   = (const float*)d_in[6];
  const float* emb1  = (const float*)d_in[7];
  const float* Wih_f = (const float*)d_in[8];
  const float* Whh_f = (const float*)d_in[9];
  const float* b_f   = (const float*)d_in[10];
  const float* Wih_b = (const float*)d_in[11];
  const float* Whh_b = (const float*)d_in[12];
  const float* b_b   = (const float*)d_in[13];
  const float* cWih  = (const float*)d_in[14];
  const float* cWhh  = (const float*)d_in[15];
  const float* cb    = (const float*)d_in[16];
  const float* f1W   = (const float*)d_in[17];
  const float* f1b   = (const float*)d_in[18];
  const float* f2W   = (const float*)d_in[19];
  const float* f2b   = (const float*)d_in[20];
  const float* f4W   = (const float*)d_in[21];
  const float* f4b   = (const float*)d_in[22];
  const float* f8W   = (const float*)d_in[23];
  const float* f8b   = (const float*)d_in[24];
  const float* imgW  = (const float*)d_in[25];
  const float* imgb  = (const float*)d_in[26];
  const float* fcW   = (const float*)d_in[27];
  const float* fcb   = (const float*)d_in[28];
  const float* fc1W  = (const float*)d_in[29];
  const float* fc1b  = (const float*)d_in[30];
  const float* pW    = (const float*)d_in[31];
  const float* pb    = (const float*)d_in[32];
  const float* a1eW  = (const float*)d_in[33];
  const float* a1eb  = (const float*)d_in[34];
  const float* a1dW  = (const float*)d_in[35];
  const float* a1db  = (const float*)d_in[36];
  const float* a1fW  = (const float*)d_in[37];
  const float* a1fb  = (const float*)d_in[38];
  const float* a2eW  = (const float*)d_in[39];
  const float* a2eb  = (const float*)d_in[40];
  const float* a2dW  = (const float*)d_in[41];
  const float* a2db  = (const float*)d_in[42];
  const float* a2fW  = (const float*)d_in[43];
  const float* a2fb  = (const float*)d_in[44];

  float* out = (float*)d_out;

  // ---- bump allocator over workspace ----
  char* wsp = (char*)d_ws;
  size_t off = 0;
  auto alloc = [&](size_t bytes) -> void* {
    void* p = wsp + off;
    off += (bytes + 255) & ~(size_t)255;
    return p;
  };
  const long NROW = (long)Bc * LARTc;          // 11520
  const long NIMG = (long)Bc * NPIXc;          // 3136

  bf16_t* Wihf_bf = (bf16_t*)alloc(2048l * 512 * 2);
  bf16_t* Whhf_bf = (bf16_t*)alloc(2048l * 512 * 2);
  bf16_t* Wihb_bf = (bf16_t*)alloc(2048l * 512 * 2);
  bf16_t* Whhb_bf = (bf16_t*)alloc(2048l * 512 * 2);
  bf16_t* cWih_bf = (bf16_t*)alloc(2048l * 1024 * 2);
  bf16_t* cWhh_bf = (bf16_t*)alloc(2048l * 512 * 2);
  bf16_t* f1W_bf  = (bf16_t*)alloc(512l * 1024 * 2);
  bf16_t* f2W_bf  = (bf16_t*)alloc(512l * 1024 * 2);
  bf16_t* f4W_bf  = (bf16_t*)alloc(512l * 1024 * 2);
  bf16_t* f8W_bf  = (bf16_t*)alloc(512l * 1024 * 2);
  bf16_t* imgW_bf = (bf16_t*)alloc(512l * 2048 * 2);
  bf16_t* fcW_bf  = (bf16_t*)alloc((long)Vc * 512 * 2);
  bf16_t* fc1W_bf = (bf16_t*)alloc(512l * 1024 * 2);
  bf16_t* a1eW_bf = (bf16_t*)alloc(512l * 512 * 2);
  bf16_t* a1dW_bf = (bf16_t*)alloc(512l * 512 * 2);
  bf16_t* a2eW_bf = (bf16_t*)alloc(512l * 512 * 2);
  bf16_t* a2dW_bf = (bf16_t*)alloc(512l * 512 * 2);
  bf16_t* enc_bf  = (bf16_t*)alloc(NIMG * 2048 * 2);
  bf16_t* E1_bf   = (bf16_t*)alloc(NROW * 512 * 2);
  float*  embs_f  = (float*)alloc((long)Bc * Tc * 512 * 4);
  float*  XWf     = (float*)alloc(NROW * 2048 * 4);
  float*  XWb     = (float*)alloc(NROW * 2048 * 4);
  bf16_t* hiddens_bf = (bf16_t*)alloc(NROW * 1024 * 2);
  bf16_t* hid1_bf = (bf16_t*)alloc(NROW * 512 * 2);
  bf16_t* hid2_bf = (bf16_t*)alloc(NROW * 512 * 2);
  bf16_t* enc1_bf = (bf16_t*)alloc(NIMG * 512 * 2);
  float*  a1encH  = (float*)alloc(NROW * 512 * 4);
  float*  a1encI  = (float*)alloc(NIMG * 512 * 4);
  float*  a2enc   = (float*)alloc(NROW * 512 * 4);
  bf16_t* hb0 = (bf16_t*)alloc(Bc * 512 * 2);
  bf16_t* hb1 = (bf16_t*)alloc(Bc * 512 * 2);
  float*  cb0 = (float*)alloc(Bc * 512 * 4);
  float*  cb1 = (float*)alloc(Bc * 512 * 4);
  bf16_t* hatt_bf = (bf16_t*)alloc(Bc * 512 * 2);
  float*  catt    = (float*)alloc(Bc * 512 * 4);
  bf16_t* f1in    = (bf16_t*)alloc(Bc * 1024 * 2);
  bf16_t* f2in    = (bf16_t*)alloc(Bc * 1024 * 2);
  bf16_t* cellin  = (bf16_t*)alloc(Bc * 1024 * 2);
  bf16_t* fc1in   = (bf16_t*)alloc(Bc * 1024 * 2);
  bf16_t* h10_bf  = (bf16_t*)alloc(Bc * 512 * 2);
  float*  a1dec   = (float*)alloc(Bc * 512 * 4);
  float*  a2dec   = (float*)alloc(Bc * 512 * 4);
  float*  s1buf   = (float*)alloc(Bc * 229 * 4);
  float*  s2buf   = (float*)alloc(Bc * 180 * 4);
  float*  betab   = (float*)alloc(Bc * 180 * 4);
  float*  guidef  = (float*)alloc(Bc * 512 * 4);
  float*  logits  = (float*)alloc((long)Bc * Vc * 4);
  float*  pg      = (float*)alloc(Bc * 4);
  float*  lsm     = (float*)alloc(Bc * 4);

  auto cvt = [&](const float* src, bf16_t* dst, long n) {
    long blocks = (n + 255) / 256;
    cvt_kernel<<<dim3((unsigned)blocks), dim3(256), 0, stream>>>(src, dst, n);
  };
  auto zero = [&](void* p, long words) {
    long blocks = (words + 255) / 256;
    zero_kernel<<<dim3((unsigned)blocks), dim3(256), 0, stream>>>((unsigned int*)p, words);
  };

  // ---- phase A: precompute ----
  cvt(Wih_f, Wihf_bf, 2048l * 512);
  cvt(Whh_f, Whhf_bf, 2048l * 512);
  cvt(Wih_b, Wihb_bf, 2048l * 512);
  cvt(Whh_b, Whhb_bf, 2048l * 512);
  cvt(cWih,  cWih_bf, 2048l * 1024);
  cvt(cWhh,  cWhh_bf, 2048l * 512);
  cvt(f1W,   f1W_bf,  512l * 1024);
  cvt(f2W,   f2W_bf,  512l * 1024);
  cvt(f4W,   f4W_bf,  512l * 1024);
  cvt(f8W,   f8W_bf,  512l * 1024);
  cvt(imgW,  imgW_bf, 512l * 2048);
  cvt(fcW,   fcW_bf,  (long)Vc * 512);
  cvt(fc1W,  fc1W_bf, 512l * 1024);
  cvt(a1eW,  a1eW_bf, 512l * 512);
  cvt(a1dW,  a1dW_bf, 512l * 512);
  cvt(a2eW,  a2eW_bf, 512l * 512);
  cvt(a2dW,  a2dW_bf, 512l * 512);
  cvt(encoder_out, enc_bf, NIMG * 2048);

  {
    long n = NROW * 512;
    gather_emb1_kernel<<<dim3((unsigned)((n + 255) / 256)), dim3(256), 0, stream>>>(emb1, articles, E1_bf, n);
    long m = (long)Bc * Tc * 512;
    gather_embs_kernel<<<dim3((unsigned)((m + 255) / 256)), dim3(256), 0, stream>>>(emb, captions, embs_f, m);
  }

  // input projections for both LSTM directions (bias folded in)
  gemm(stream, E1_bf, 512, Wihf_bf, 512, b_f, XWf, 2048, nullptr, 0, (int)NROW, 2048, 512, 0);
  gemm(stream, E1_bf, 512, Wihb_bf, 512, b_b, XWb, 2048, nullptr, 0, (int)NROW, 2048, 512, 0);

  // forward LSTM scan
  zero(hb0, Bc * 512 / 2);
  zero(cb0, Bc * 512);
  for (int t = 0; t < LARTc; ++t) {
    const bf16_t* hs = (t & 1) ? hb1 : hb0;
    const float*  cs = (t & 1) ? cb1 : cb0;
    bf16_t* hd = (t & 1) ? hb0 : hb1;
    float*  cd = (t & 1) ? cb0 : cb1;
    lstm_step_kernel<<<dim3(16), dim3(256), 0, stream>>>(hs, cs, Whhf_bf, XWf, t, LARTc, cd, hd, hiddens_bf, 0);
  }
  // backward LSTM scan
  zero(hb0, Bc * 512 / 2);
  zero(cb0, Bc * 512);
  for (int i = 0; i < LARTc; ++i) {
    int t = LARTc - 1 - i;
    const bf16_t* hs = (i & 1) ? hb1 : hb0;
    const float*  cs = (i & 1) ? cb1 : cb0;
    bf16_t* hd = (i & 1) ? hb0 : hb1;
    float*  cd = (i & 1) ? cb0 : cb1;
    lstm_step_kernel<<<dim3(16), dim3(256), 0, stream>>>(hs, cs, Whhb_bf, XWb, t, LARTc, cd, hd, hiddens_bf, 512);
  }

  // hiddens1 / hiddens2 / enc1 / attention encoder projections
  gemm(stream, hiddens_bf, 1024, f4W_bf, 1024, f4b, nullptr, 0, hid1_bf, 512, (int)NROW, 512, 1024, 1);
  gemm(stream, hiddens_bf, 1024, f8W_bf, 1024, f8b, nullptr, 0, hid2_bf, 512, (int)NROW, 512, 1024, 1);
  gemm(stream, enc_bf, 2048, imgW_bf, 2048, imgb, nullptr, 0, enc1_bf, 512, (int)NIMG, 512, 2048, 1);
  gemm(stream, hid1_bf, 512, a1eW_bf, 512, a1eb, a1encH, 512, nullptr, 0, (int)NROW, 512, 512, 0);
  gemm(stream, enc1_bf, 512, a1eW_bf, 512, a1eb, a1encI, 512, nullptr, 0, (int)NIMG, 512, 512, 0);
  gemm(stream, hid2_bf, 512, a2eW_bf, 512, a2eb, a2enc, 512, nullptr, 0, (int)NROW, 512, 512, 0);

  // decoder state init
  zero(hatt_bf, Bc * 512 / 2);
  zero(catt, Bc * 512);
  zero(f1in, Bc * 1024 / 2);
  zero(f2in, Bc * 1024 / 2);

  // ---- phase B: decoder loop ----
  for (int t = 0; t < Tc; ++t) {
    gemm(stream, hatt_bf, 512, a1dW_bf, 512, a1db, a1dec, 512, nullptr, 0, Bc, 512, 512, 0);
    score1_kernel<<<dim3((Bc * 229 + 7) / 8), dim3(256), 0, stream>>>(a1encH, a1encI, a1dec, a1fW, a1fb, s1buf, Bc * 229);
    att1_kernel<<<dim3(Bc), dim3(256), 0, stream>>>(s1buf, hid1_bf, enc1_bf, embs_f, t, cellin);
    shift_kernel<<<dim3(128), dim3(256), 0, stream>>>(f1in, f2in);
    cell_step_kernel<<<dim3(16), dim3(256), 0, stream>>>(cellin, hatt_bf, cWih_bf, cWhh_bf, cb, catt, f1in, f2in);
    gemm(stream, f1in, 1024, a2dW_bf, 512, a2db, a2dec, 512, nullptr, 0, Bc, 512, 512, 0);
    score2_kernel<<<dim3((Bc * 180 + 7) / 8), dim3(256), 0, stream>>>(a2enc, a2dec, a2fW, a2fb, mask, s2buf, Bc * 180);
    att2_kernel<<<dim3(Bc), dim3(256), 0, stream>>>(s2buf, hid2_bf, f1in, betab, guidef, fc1in);
    pgate_kernel<<<dim3(8), dim3(256), 0, stream>>>(embs_f, guidef, f1in, pW, pb, t, pg);
    gemm(stream, fc1in, 1024, fc1W_bf, 1024, fc1b, nullptr, 0, h10_bf, 512, Bc, 512, 1024, 2);
    gemm(stream, h10_bf, 512, fcW_bf, 512, fcb, logits, Vc, nullptr, 0, Bc, Vc, 512, 0);
    lse_kernel<<<dim3(Bc), dim3(256), 0, stream>>>(logits, lsm);
    {
      long n = (long)Bc * Vc;
      final_kernel<<<dim3((unsigned)((n + 255) / 256)), dim3(256), 0, stream>>>(logits, pg, lsm, caplen, t, out);
    }
    scatter_kernel<<<dim3((Bc * LARTc + 255) / 256), dim3(256), 0, stream>>>(betab, pg, articles, caplen, t, out);
    gemm(stream, f1in, 1024, f1W_bf, 1024, f1b, nullptr, 0, hatt_bf, 512, Bc, 512, 1024, 1);
    gemm(stream, f2in, 1024, f2W_bf, 1024, f2b, catt, 512, nullptr, 0, Bc, 512, 1024, 1);
  }
}
